// AutoEncoder_v2_74345883894285
// MI455X (gfx1250) — matrix operand order repaired and checked by reading, not compiled, not run on hardware
//
#include <hip/hip_runtime.h>

// ---------------------------------------------------------------------------
// CDNA5 (gfx1250) fused autoencoder pipeline.
// All GEMMs use V_WMMA_F32_16X16X32_F16 (wave32). Weights are pre-swizzled
// into per-lane B-fragment order so each lane fetches its 32B v16h fragment
// with one contiguous global load (L2-resident). Activations are staged in
// LDS as f16; each of the 8 waves in a 256-thread block owns a 16-row tile.
// ---------------------------------------------------------------------------

typedef _Float16 half_t;
typedef __attribute__((ext_vector_type(16))) _Float16 v16h;
typedef __attribute__((ext_vector_type(8)))  _Float16 v8h;
typedef __attribute__((ext_vector_type(8)))  float    v8f;

static_assert(sizeof(v16h) == 32, "v16h must be 32 bytes");

constexpr int BATCH = 65536;
constexpr int ROWS  = 128;       // rows per workgroup
constexpr int NBLK  = BATCH / ROWS;   // 512
constexpr int STR   = 264;       // LDS activation stride (halfs), 16B-aligned

// ---- workspace layout (bytes) ----
constexpr size_t OFF_X    = 0;                               // x buffer: B*16 f32
constexpr size_t OFF_PART = (size_t)BATCH * 16 * 4;          // 4 MiB
constexpr size_t OFF_SCAL = OFF_PART + 4096;                 // partials: 512 f32
constexpr size_t OFF_W    = OFF_SCAL + 256;                  // swizzled f16 weights

// half-element offsets within the swizzled weight arena
constexpr int W_pW1 = 0;                     // 160x128: 5*8  tiles
constexpr int W_pW2 = W_pW1 + 5 * 8 * 512;   // 128x256: 4*16
constexpr int W_pW3 = W_pW2 + 4 * 16 * 512;  // 256x256: 8*16
constexpr int W_pW4 = W_pW3 + 8 * 16 * 512;  // 256x128: 8*8
constexpr int W_eW1 = W_pW4 + 8 * 8 * 512;   // 256x128: 8*8
constexpr int W_eW2 = W_eW1 + 8 * 8 * 512;   // 128x16 : 4*1
constexpr int W_dW1 = W_eW2 + 4 * 1 * 512;   // 32x256 (K padded 16->32): 1*16
constexpr int W_dW2 = W_dW1 + 1 * 16 * 512;  // 256x256: 8*16

// ---------------------------------------------------------------------------
// Phase 0: swizzle W (f32, K x N row-major) -> f16 fragments.
// Tile (kt, nt): 32 lanes x 16 halfs; lane L element e holds
//   W[kt*32 + (L/16)*16 + e][nt*16 + L%16]   (zero if k >= K, for K padding)
// Tile order: kt fastest within nt (so a wave's k-loop streams 1KB tiles).
// ---------------------------------------------------------------------------
__global__ __launch_bounds__(256) void swizzle_w_kernel(
    const float* __restrict__ W, half_t* __restrict__ dst, int K, int Kp, int N) {
  int tid    = blockIdx.x * 256 + threadIdx.x;
  int ktiles = Kp / 32;
  int tiles  = ktiles * (N / 16);
  if (tid >= tiles * 32) return;
  int lane = tid & 31;
  int tile = tid >> 5;
  int kt   = tile % ktiles;
  int nt   = tile / ktiles;
  int n     = nt * 16 + (lane & 15);
  int kbase = kt * 32 + (lane >> 4) * 16;
  v16h out;
#pragma unroll
  for (int e = 0; e < 16; ++e) {
    int k = kbase + e;
    float f = (k < K) ? W[(size_t)k * N + n] : 0.0f;
    out[e] = (half_t)f;
  }
  *(v16h*)(dst + (size_t)tile * 512 + lane * 16) = out;
}

// ---------------------------------------------------------------------------
// Wave-level GEMM helpers.  Each wave owns rows [wave*16, wave*16+16) of the
// LDS activation tile.  A-fragments (v16h) are hoisted across the ntile loop.
// Layout used here (A/B k order is contiguous, not the ISA table order):
//   A: m = lane%16;  k = kt*32 + (lane/16)*16 + e
//   B: n = lane%16;  k = kt*32 + (lane/16)*16 + e   (pre-swizzled in global)
//   D: row = v + 8*(lane/16); col = nt*16 + lane%16
// ---------------------------------------------------------------------------
template <int KT>
__device__ __forceinline__ void load_afrags(const half_t* __restrict__ xrow,
                                            v16h (&af)[KT], int hs) {
#pragma unroll
  for (int kt = 0; kt < KT; ++kt) {
    int k0 = kt * 32 + hs * 16;
    v8h lo = *(const v8h*)(xrow + k0);
    v8h hi = *(const v8h*)(xrow + k0 + 8);
#pragma unroll
    for (int i = 0; i < 8; ++i) { af[kt][i] = lo[i]; af[kt][i + 8] = hi[i]; }
  }
}

template <int KT, int NT, bool RELU>
__device__ __forceinline__ void mlp_layer(const half_t* __restrict__ Xs, int xstride,
                                          half_t* __restrict__ Ys, int ystride,
                                          const half_t* __restrict__ Wsw,
                                          const float* __restrict__ bias,
                                          int lane, int wave) {
  const int m = lane & 15, hs = lane >> 4;
  const half_t* xrow = Xs + (size_t)(wave * 16 + m) * xstride;
  v16h af[KT];
  load_afrags<KT>(xrow, af, hs);
  for (int nt = 0; nt < NT; ++nt) {
    float bv = bias[nt * 16 + m];
    v8f acc;
#pragma unroll
    for (int v = 0; v < 8; ++v) acc[v] = bv;
    const half_t* wt = Wsw + (size_t)nt * KT * 512 + lane * 16;
#pragma unroll
    for (int kt = 0; kt < KT; ++kt) {
      v16h b = *(const v16h*)(wt + (size_t)kt * 512);
      acc = __builtin_amdgcn_wmma_f32_16x16x32_f16(false, af[kt], false, b,
                                                   (short)0, acc, false, false);
    }
#pragma unroll
    for (int v = 0; v < 8; ++v) {
      float f = acc[v];
      if (RELU) f = fmaxf(f, 0.0f);
      Ys[(size_t)(wave * 16 + v + 8 * hs) * ystride + nt * 16 + m] = (half_t)f;
    }
  }
}

// Variant: write f32 to global and accumulate per-row sum of squares -> rowN2.
template <int KT, int NT>
__device__ __forceinline__ void mlp_layer_gout(const half_t* __restrict__ Xs, int xstride,
                                               float* __restrict__ gdst, int gstride, int row0g,
                                               const half_t* __restrict__ Wsw,
                                               const float* __restrict__ bias,
                                               float* __restrict__ rowN2,
                                               int lane, int wave) {
  const int m = lane & 15, hs = lane >> 4;
  const half_t* xrow = Xs + (size_t)(wave * 16 + m) * xstride;
  v16h af[KT];
  load_afrags<KT>(xrow, af, hs);
  float h2[8];
#pragma unroll
  for (int v = 0; v < 8; ++v) h2[v] = 0.0f;
  for (int nt = 0; nt < NT; ++nt) {
    float bv = bias[nt * 16 + m];
    v8f acc;
#pragma unroll
    for (int v = 0; v < 8; ++v) acc[v] = bv;
    const half_t* wt = Wsw + (size_t)nt * KT * 512 + lane * 16;
#pragma unroll
    for (int kt = 0; kt < KT; ++kt) {
      v16h b = *(const v16h*)(wt + (size_t)kt * 512);
      acc = __builtin_amdgcn_wmma_f32_16x16x32_f16(false, af[kt], false, b,
                                                   (short)0, acc, false, false);
    }
#pragma unroll
    for (int v = 0; v < 8; ++v) {
      float f = acc[v];
      gdst[(size_t)(row0g + wave * 16 + v + 8 * hs) * gstride + nt * 16 + m] = f;
      h2[v] += f * f;
    }
  }
  // reduce over the 16 column-lanes of each half (masks 1,2,4,8 stay in-half)
#pragma unroll
  for (int v = 0; v < 8; ++v) {
    float s = h2[v];
    s += __shfl_xor(s, 1); s += __shfl_xor(s, 2);
    s += __shfl_xor(s, 4); s += __shfl_xor(s, 8);
    if (m == 0) rowN2[wave * 16 + v + 8 * hs] = s;
  }
}

// ---------------------------------------------------------------------------
// Phase 1: precoder (4 layers) -> hhat (output 1) + ||h||^2;
//          encoder (2 layers) -> x (workspace) + ||x||^2; block partial sum.
// ---------------------------------------------------------------------------
__global__ __launch_bounds__(256) void phase1_kernel(
    const float* __restrict__ channel, const float* __restrict__ message,
    const float* __restrict__ pb1, const float* __restrict__ pb2,
    const float* __restrict__ pb3, const float* __restrict__ pb4,
    const float* __restrict__ eb1, const float* __restrict__ eb2,
    const half_t* __restrict__ WS, float* __restrict__ xbuf,
    float* __restrict__ partials, float* __restrict__ hhat_out) {
  __shared__ __attribute__((aligned(16))) half_t A0[ROWS * STR];
  __shared__ __attribute__((aligned(16))) half_t A1[ROWS * STR];
  __shared__ float rowH[ROWS];
  __shared__ float rowX[ROWS];
  __shared__ float red[ROWS];

  const int t = threadIdx.x, lane = t & 31, wave = t >> 5;
  const int r0g = blockIdx.x * ROWS;

  // stage channel (128 x 160 f32 -> f16)
  for (int idx = t; idx < ROWS * 160; idx += 256) {
    int r = idx / 160, c = idx % 160;
    A0[r * STR + c] = (half_t)channel[(size_t)(r0g + r) * 160 + c];
  }
  __syncthreads();
  mlp_layer<5, 8, true>(A0, STR, A1, STR, WS + W_pW1, pb1, lane, wave);   // 160->128
  __syncthreads();
  mlp_layer<4, 16, true>(A1, STR, A0, STR, WS + W_pW2, pb2, lane, wave);  // 128->256
  __syncthreads();
  mlp_layer<8, 16, true>(A0, STR, A1, STR, WS + W_pW3, pb3, lane, wave);  // 256->256
  __syncthreads();
  // hhat = h @ pW4 + pb4 (no relu) -> global f32 + ||h||^2
  mlp_layer_gout<8, 8>(A1, STR, hhat_out, 128, r0g, WS + W_pW4, pb4, rowH, lane, wave);

  // stage message (A0 is free after the L3 barrier)
  for (int idx = t; idx < ROWS * 256; idx += 256) {
    int r = idx >> 8, c = idx & 255;
    A0[r * STR + c] = (half_t)message[(size_t)(r0g + r) * 256 + c];
  }
  __syncthreads();
  mlp_layer<8, 8, true>(A0, STR, A1, STR, WS + W_eW1, eb1, lane, wave);   // 256->128
  __syncthreads();
  mlp_layer_gout<4, 1>(A1, STR, xbuf, 16, r0g, WS + W_eW2, eb2, rowX, lane, wave); // 128->16
  __syncthreads();

  // block partial of sum_b ||x||^2 * ||h||^2   (deterministic, no atomics)
  if (t < ROWS) red[t] = rowH[t] * rowX[t];
  __syncthreads();
  for (int s = ROWS / 2; s > 0; s >>= 1) {
    if (t < s) red[t] += red[t + s];
    __syncthreads();
  }
  if (t == 0) partials[blockIdx.x] = red[0];
}

// ---------------------------------------------------------------------------
// Phase 1.5: reduce 512 partials -> scal[0] = 1/norms
// ---------------------------------------------------------------------------
__global__ __launch_bounds__(512) void reduce_norm_kernel(
    const float* __restrict__ partials, float* __restrict__ scal) {
  __shared__ float red[512];
  int t = threadIdx.x;
  red[t] = partials[t];
  __syncthreads();
  for (int s = 256; s > 0; s >>= 1) {
    if (t < s) red[t] += red[t + s];
    __syncthreads();
  }
  if (t == 0) {
    float S = red[0];
    float norms = sqrtf(S / (float)((double)BATCH * 8.0));
    scal[0] = 1.0f / norms;
  }
}

// ---------------------------------------------------------------------------
// Deterministic noise matching jax.random.normal(key(42), (B,16)):
// threefry2x32 (20 rounds) over the iota split-halves, then the
// uniform(-1,1)->sqrt(2)*erfinv(u) transform (Giles erfinv approximation).
// ---------------------------------------------------------------------------
__device__ __forceinline__ unsigned rotl32(unsigned x, unsigned r) {
  return (x << r) | (x >> (32u - r));
}

__device__ float jax_normal42(unsigned idx) {
  const unsigned HALF_N = (unsigned)(BATCH * 16 / 2);  // 524288
  unsigned x0, x1;
  bool second;
  if (idx < HALF_N) { x0 = idx; x1 = idx + HALF_N; second = false; }
  else              { x0 = idx - HALF_N; x1 = idx; second = true; }
  const unsigned ks0 = 0u, ks1 = 42u, ks2 = 0x1BD11BDAu ^ ks0 ^ ks1;
  const unsigned rotA[4] = {13u, 15u, 26u, 6u};
  const unsigned rotB[4] = {17u, 29u, 16u, 24u};
  const unsigned ks[3] = {ks0, ks1, ks2};
  x0 += ks0; x1 += ks1;
#pragma unroll
  for (int g = 0; g < 5; ++g) {
#pragma unroll
    for (int r = 0; r < 4; ++r) {
      unsigned rot = (g & 1) ? rotB[r] : rotA[r];
      x0 += x1;
      x1 = rotl32(x1, rot) ^ x0;
    }
    x0 += ks[(g + 1) % 3];
    x1 += ks[(g + 2) % 3] + (unsigned)(g + 1);
  }
  unsigned bits = second ? x1 : x0;
  float u01 = __uint_as_float((bits >> 9) | 0x3f800000u) - 1.0f;  // [0,1)
  const float lo = -0.99999994f;                                  // nextafter(-1,0)
  float u = u01 * (1.0f - lo) + lo;
  u = fmaxf(u, lo);
  // erfinv (single-precision polynomial, Giles 2012)
  float w = -logf((1.0f - u) * (1.0f + u));
  float p;
  if (w < 5.0f) {
    w -= 2.5f;
    p = 2.81022636e-08f;
    p = fmaf(p, w, 3.43273939e-07f);
    p = fmaf(p, w, -3.5233877e-06f);
    p = fmaf(p, w, -4.39150654e-06f);
    p = fmaf(p, w, 0.00021858087f);
    p = fmaf(p, w, -0.00125372503f);
    p = fmaf(p, w, -0.00417768164f);
    p = fmaf(p, w, 0.246640727f);
    p = fmaf(p, w, 1.50140941f);
  } else {
    w = sqrtf(w) - 3.0f;
    p = -0.000200214257f;
    p = fmaf(p, w, 0.000100950558f);
    p = fmaf(p, w, 0.00134934322f);
    p = fmaf(p, w, -0.00367342844f);
    p = fmaf(p, w, 0.00573950773f);
    p = fmaf(p, w, -0.0076224613f);
    p = fmaf(p, w, 0.00943887047f);
    p = fmaf(p, w, 1.00167406f);
    p = fmaf(p, w, 2.83297682f);
  }
  return 1.41421356237f * (p * u);
}

// ---------------------------------------------------------------------------
// Phase 2: P/Q dots, receive + noise, decoder (2 WMMA layers), softmax.
// ---------------------------------------------------------------------------
__global__ __launch_bounds__(256) void phase2_kernel(
    const float* __restrict__ hhat, const float* __restrict__ dlink,
    const float* __restrict__ noise_std, const float* __restrict__ xbuf,
    const float* __restrict__ scal,
    const float* __restrict__ db1, const float* __restrict__ db2,
    const half_t* __restrict__ WS, float* __restrict__ dec_out) {
  __shared__ float Ps[ROWS];
  __shared__ float Qs[ROWS];
  __shared__ __attribute__((aligned(16))) half_t rbuf[ROWS * 32];   // K padded to 32
  __shared__ __attribute__((aligned(16))) half_t tbuf[ROWS * STR];
  __shared__ __attribute__((aligned(16))) half_t lbuf[ROWS * STR];  // f16 logits

  const int t = threadIdx.x, lane = t & 31, wave = t >> 5;
  const int r0g = blockIdx.x * ROWS;
  const float invn = scal[0];
  const float nstd = noise_std[0];

  // P = dr.hr - di.hi ; Q = dr.hi + di.hr   (wave-per-row, coalesced 128B)
  for (int i = 0; i < 16; ++i) {
    int rl = wave * 16 + i;
    const float* hh = hhat  + (size_t)(r0g + rl) * 128;
    const float* dl = dlink + (size_t)(r0g + rl) * 128;
    float hrA = hh[lane],      hrB = hh[32 + lane];
    float hiA = hh[64 + lane], hiB = hh[96 + lane];
    float drA = dl[lane],      drB = dl[32 + lane];
    float diA = dl[64 + lane], diB = dl[96 + lane];
    float Drr = hrA * drA + hrB * drB;
    float Dri = hiA * drA + hiB * drB;
    float Dir = hrA * diA + hrB * diB;
    float Dii = hiA * diA + hiB * diB;
#pragma unroll
    for (int m = 1; m < 32; m <<= 1) {
      Drr += __shfl_xor(Drr, m); Dri += __shfl_xor(Dri, m);
      Dir += __shfl_xor(Dir, m); Dii += __shfl_xor(Dii, m);
    }
    if (lane == 0) { Ps[rl] = Drr - Dii; Qs[rl] = Dri + Dir; }
  }
  __syncthreads();

  // zero-pad r tile, then fill received symbols + noise
  for (int idx = t; idx < ROWS * 32; idx += 256) rbuf[idx] = (half_t)0.0f;
  __syncthreads();
  for (int p = t; p < ROWS * 8; p += 256) {
    int rl = p >> 3, n = p & 7;
    int rg = r0g + rl;
    float xr = xbuf[(size_t)rg * 16 + 2 * n];
    float xi = xbuf[(size_t)rg * 16 + 2 * n + 1];
    float P = Ps[rl], Q = Qs[rl];
    float rx = (xr * P - xi * Q) * invn;
    float ry = (xr * Q + xi * P) * invn;
    unsigned i0 = (unsigned)(rg * 16 + 2 * n);
    rx += nstd * jax_normal42(i0);
    ry += nstd * jax_normal42(i0 + 1u);
    rbuf[rl * 32 + 2 * n]     = (half_t)rx;
    rbuf[rl * 32 + 2 * n + 1] = (half_t)ry;
  }
  __syncthreads();

  mlp_layer<1, 16, true>(rbuf, 32, tbuf, STR, WS + W_dW1, db1, lane, wave);   // 16(32)->256
  __syncthreads();
  mlp_layer<8, 16, false>(tbuf, STR, lbuf, STR, WS + W_dW2, db2, lane, wave); // 256->256
  __syncthreads();

  // softmax over 256 cols (wave-per-row; lane covers cols l, l+32, ... l+224)
  for (int i = 0; i < 16; ++i) {
    int rl = wave * 16 + i;
    int rg = r0g + rl;
    float v[8];
    float mx = -3.0e38f;
#pragma unroll
    for (int j = 0; j < 8; ++j) {
      v[j] = (float)lbuf[rl * STR + lane + 32 * j];
      mx = fmaxf(mx, v[j]);
    }
#pragma unroll
    for (int m = 1; m < 32; m <<= 1) mx = fmaxf(mx, __shfl_xor(mx, m));
    float s = 0.0f;
#pragma unroll
    for (int j = 0; j < 8; ++j) { v[j] = expf(v[j] - mx); s += v[j]; }
#pragma unroll
    for (int m = 1; m < 32; m <<= 1) s += __shfl_xor(s, m);
    float inv = 1.0f / s;
#pragma unroll
    for (int j = 0; j < 8; ++j)
      dec_out[(size_t)rg * 256 + lane + 32 * j] = v[j] * inv;
  }
}

// ---------------------------------------------------------------------------
// Host launch. d_in order: channel, message, downlink_data, noise_std, hhat,
// pW1, pb1, pW2, pb2, pW3, pb3, pW4, pb4, eW1, eb1, eW2, eb2, dW1, db1, dW2, db2
// d_out: [hhat (B*128 f32) | h_dec (B*256 f32)]
// ---------------------------------------------------------------------------
extern "C" void kernel_launch(void* const* d_in, const int* in_sizes, int n_in,
                              void* d_out, int out_size, void* d_ws, size_t ws_size,
                              hipStream_t stream) {
  (void)in_sizes; (void)n_in; (void)out_size; (void)ws_size;

  const float* channel = (const float*)d_in[0];
  const float* message = (const float*)d_in[1];
  const float* dlink   = (const float*)d_in[2];
  const float* nstd    = (const float*)d_in[3];
  const float* pW1 = (const float*)d_in[5];  const float* pb1 = (const float*)d_in[6];
  const float* pW2 = (const float*)d_in[7];  const float* pb2 = (const float*)d_in[8];
  const float* pW3 = (const float*)d_in[9];  const float* pb3 = (const float*)d_in[10];
  const float* pW4 = (const float*)d_in[11]; const float* pb4 = (const float*)d_in[12];
  const float* eW1 = (const float*)d_in[13]; const float* eb1 = (const float*)d_in[14];
  const float* eW2 = (const float*)d_in[15]; const float* eb2 = (const float*)d_in[16];
  const float* dW1 = (const float*)d_in[17]; const float* db1 = (const float*)d_in[18];
  const float* dW2 = (const float*)d_in[19]; const float* db2 = (const float*)d_in[20];

  char* ws = (char*)d_ws;
  float*  xbuf     = (float*)(ws + OFF_X);
  float*  partials = (float*)(ws + OFF_PART);
  float*  scal     = (float*)(ws + OFF_SCAL);
  half_t* WS       = (half_t*)(ws + OFF_W);

  float* hhat_out = (float*)d_out;                       // B x 128
  float* dec_out  = (float*)d_out + (size_t)BATCH * 128; // B x 256

  auto sw = [&](const float* W, half_t* dst, int K, int Kp, int N) {
    int threads = (Kp / 32) * (N / 16) * 32;
    swizzle_w_kernel<<<(threads + 255) / 256, 256, 0, stream>>>(W, dst, K, Kp, N);
  };
  sw(pW1, WS + W_pW1, 160, 160, 128);
  sw(pW2, WS + W_pW2, 128, 128, 256);
  sw(pW3, WS + W_pW3, 256, 256, 256);
  sw(pW4, WS + W_pW4, 256, 256, 128);
  sw(eW1, WS + W_eW1, 256, 256, 128);
  sw(eW2, WS + W_eW2, 128, 128, 16);
  sw(dW1, WS + W_dW1, 16, 32, 256);   // pad K 16 -> 32
  sw(dW2, WS + W_dW2, 256, 256, 256);

  phase1_kernel<<<NBLK, 256, 0, stream>>>(channel, message, pb1, pb2, pb3, pb4,
                                          eb1, eb2, WS, xbuf, partials, hhat_out);
  reduce_norm_kernel<<<1, 512, 0, stream>>>(partials, scal);
  phase2_kernel<<<NBLK, 256, 0, stream>>>(hhat_out, dlink, nstd, xbuf, scal,
                                          db1, db2, WS, dec_out);
}
